// DeepIRT_20177756357259
// MI455X (gfx1250) — matrix operand order repaired and checked by reading, not compiled, not run on hardware
//
#include <hip/hip_runtime.h>
#include <hip/hip_bf16.h>

// ---------------------------------------------------------------------------
// Deep-IRT / DKVMN inference for MI455X (gfx1250, wave32, WMMA bf16).
//   B=64, T=500, D=128, M=50, rows = 32000.
// Phase 1: convert tables/weights to bf16 (weights transposed for B-frag loads)
// Phase 2: per-row gather + WMMA GEMMs -> w (softmax), e (sigmoid), a (tanh)
// Phase 3: serial DKVMN scan, Mv column register-resident, 64 blocks
// Phase 4: WMMA GEMM f = tanh([read,k]@Wf) + fused prediction epilogue
// ---------------------------------------------------------------------------

typedef unsigned short u16;
typedef __attribute__((ext_vector_type(8)))  u16    u16x8;
typedef __attribute__((ext_vector_type(16))) u16    u16x16;
typedef __attribute__((ext_vector_type(16))) __bf16 bf16x16;
typedef __attribute__((ext_vector_type(8)))  float  f32x8;

#define NUM_CONCEPT 1000
#define DIM   128
#define MEM   50
#define BATCH 64
#define SEQT  500
#define ROWS  (BATCH * SEQT)      // 32000
#define TILES (ROWS / 16)         // 2000
#define WAVES_PER_BLOCK 4

// ---- workspace layout (bytes), all offsets multiples of 256 ----
#define OFF_EK   0u                     // 1000*128 bf16      = 256000
#define OFF_EV   256000u                // 2000*128 bf16      = 512000
#define OFF_MK   768000u                // 64*128 bf16 (pad)  = 16384
#define OFF_WET  784384u                // WeT [128][128]     = 32768
#define OFF_WAT  817152u                // WaT [128][128]     = 32768
#define OFF_WFT  849920u                // WfT [128][256]     = 65536
#define OFF_KBF  915456u                // k bf16 [32000][128]= 8192000
#define OFF_W    9107456u               // w f32 [32000][64]  = 8192000
#define OFF_E    17299456u              // e f32 [32000][128] = 16384000
#define OFF_A    33683456u              // a f32 [32000][128] = 16384000
#define OFF_RD   50067456u              // read bf16 [32000][128] = 8192000

__device__ __forceinline__ u16 f2bf(float f) {
    unsigned u = __builtin_bit_cast(unsigned, f);
    unsigned r = (u + 0x7FFFu + ((u >> 16) & 1u)) >> 16;   // RNE
    return (u16)r;
}
__device__ __forceinline__ float bf2f(u16 h) {
    unsigned u = ((unsigned)h) << 16;
    return __builtin_bit_cast(float, u);
}
__device__ __forceinline__ bf16x16 make_frag(const u16* p0, const u16* p1) {
    u16x8 lo = *(const u16x8*)p0;     // 16B contiguous
    u16x8 hi = *(const u16x8*)p1;     // 16B contiguous
    u16x16 c = __builtin_shufflevector(lo, hi, 0,1,2,3,4,5,6,7,8,9,10,11,12,13,14,15);
    return __builtin_bit_cast(bf16x16, c);
}
__device__ __forceinline__ f32x8 zero8() {
    f32x8 v;
#pragma unroll
    for (int i = 0; i < 8; ++i) v[i] = 0.0f;
    return v;
}
__device__ __forceinline__ float sigmoidf(float x) {
    return 1.0f / (1.0f + __expf(-x));
}

// ---------------------------------------------------------------------------
// Phase 1: bf16 conversion kernels
// ---------------------------------------------------------------------------
__global__ void conv_copy(const float* __restrict__ src, u16* __restrict__ dst, int n) {
    int i = blockIdx.x * blockDim.x + threadIdx.x;
    if (i < n) dst[i] = f2bf(src[i]);
}
// Mk [50][128] -> padded [64][128] bf16 (rows >=50 zero)
__global__ void conv_mk(const float* __restrict__ Mk, u16* __restrict__ dst) {
    int i = blockIdx.x * blockDim.x + threadIdx.x;   // 0..8191
    int m = i >> 7, c = i & 127;
    dst[i] = (m < MEM) ? f2bf(Mk[m * DIM + c]) : (u16)0;
}
// src [K][N] row-major f32 -> dst [N][K] bf16 (so B-frags are K-contiguous)
__global__ void conv_T(const float* __restrict__ src, u16* __restrict__ dst, int K, int N) {
    int i = blockIdx.x * blockDim.x + threadIdx.x;
    if (i < K * N) {
        int n = i / K, k = i % K;
        dst[n * K + k] = f2bf(src[k * N + n]);
    }
}

// ---------------------------------------------------------------------------
// Phase 2: gather + WMMA: w = softmax(k@Mk^T), e = sigmoid(v@We+be),
//          a = tanh(v@Wa+ba). One wave per 16-row tile, 4 waves per block.
// ---------------------------------------------------------------------------
__global__ __launch_bounds__(128) void dkvmn_embed_gemm(
    const int* __restrict__ concept_seq, const int* __restrict__ correct_seq,
    const u16* __restrict__ ek, const u16* __restrict__ ev,
    const u16* __restrict__ mk, const u16* __restrict__ WeT,
    const u16* __restrict__ WaT,
    const float* __restrict__ be, const float* __restrict__ ba,
    u16* __restrict__ k_bf, float* __restrict__ w_out,
    float* __restrict__ e_out, float* __restrict__ a_out)
{
    __shared__ float lds[WAVES_PER_BLOCK][16][64];

    const int wave = threadIdx.x >> 5;
    const int lane = threadIdx.x & 31;
    const int tile = blockIdx.x * WAVES_PER_BLOCK + wave;   // < 2000
    const int rowBase = tile * 16;
    const int r    = lane & 15;
    const int half = lane >> 4;          // 0: K-lo group, 1: K-hi group
    const int row  = rowBase + r;
    const int koff = half * 8;

    const int ci = concept_seq[row];
    const int vi = ci + NUM_CONCEPT * correct_seq[row];
    const u16* krow = ek + ci * DIM;
    const u16* vrow = ev + vi * DIM;

    // Build A-fragments for k and v (16x32 bf16 chunks, ISA lane layout),
    // and persist k in bf16 for phase 4.
    bf16x16 kA[4], vA[4];
#pragma unroll
    for (int c = 0; c < 4; ++c) {
        int kb = c * 32 + koff;
        kA[c] = make_frag(krow + kb, krow + kb + 16);
        vA[c] = make_frag(vrow + kb, vrow + kb + 16);
        *(u16x8*)(k_bf + row * DIM + kb)      = *(const u16x8*)(krow + kb);
        *(u16x8*)(k_bf + row * DIM + kb + 16) = *(const u16x8*)(krow + kb + 16);
    }

    // ---- logits = k @ Mk^T  -> [16 x 64] (N padded 50->64) ----
    f32x8 accW[4];
#pragma unroll
    for (int n = 0; n < 4; ++n) accW[n] = zero8();
#pragma unroll
    for (int c = 0; c < 4; ++c) {
#pragma unroll
        for (int n = 0; n < 4; ++n) {
            const u16* bp = mk + (n * 16 + r) * DIM + c * 32 + half * 8;
            bf16x16 B = make_frag(bp, bp + 16);
            accW[n] = __builtin_amdgcn_wmma_f32_16x16x32_bf16(
                false, kA[c], false, B, (short)0, accW[n], false, false);
        }
    }
    // softmax over M=50 via LDS (per-row, 16 lanes)
#pragma unroll
    for (int n = 0; n < 4; ++n)
#pragma unroll
        for (int i = 0; i < 8; ++i)
            lds[wave][i + half * 8][n * 16 + r] = accW[n][i];
    __syncthreads();
    if (lane < 16) {
        float mx = -1e30f;
        for (int m = 0; m < MEM; ++m) mx = fmaxf(mx, lds[wave][r][m]);
        float s = 0.0f;
        for (int m = 0; m < MEM; ++m) s += __expf(lds[wave][r][m] - mx);
        float inv = 1.0f / s;
        for (int m = 0; m < MEM; ++m)
            w_out[row * 64 + m] = __expf(lds[wave][r][m] - mx) * inv;
    }

    // ---- e = sigmoid(v @ We + be) ----
    f32x8 acc[8];
#pragma unroll
    for (int j = 0; j < 8; ++j) acc[j] = zero8();
#pragma unroll
    for (int c = 0; c < 4; ++c) {
#pragma unroll
        for (int j = 0; j < 8; ++j) {
            const u16* bp = WeT + (j * 16 + r) * DIM + c * 32 + half * 8;
            bf16x16 B = make_frag(bp, bp + 16);
            acc[j] = __builtin_amdgcn_wmma_f32_16x16x32_bf16(
                false, vA[c], false, B, (short)0, acc[j], false, false);
        }
    }
#pragma unroll
    for (int j = 0; j < 8; ++j) {
        float bias = be[j * 16 + r];
#pragma unroll
        for (int i = 0; i < 8; ++i) {
            int m = i + half * 8;
            e_out[(rowBase + m) * DIM + j * 16 + r] = sigmoidf(acc[j][i] + bias);
        }
    }

    // ---- a = tanh(v @ Wa + ba) ----
#pragma unroll
    for (int j = 0; j < 8; ++j) acc[j] = zero8();
#pragma unroll
    for (int c = 0; c < 4; ++c) {
#pragma unroll
        for (int j = 0; j < 8; ++j) {
            const u16* bp = WaT + (j * 16 + r) * DIM + c * 32 + half * 8;
            bf16x16 B = make_frag(bp, bp + 16);
            acc[j] = __builtin_amdgcn_wmma_f32_16x16x32_bf16(
                false, vA[c], false, B, (short)0, acc[j], false, false);
        }
    }
#pragma unroll
    for (int j = 0; j < 8; ++j) {
        float bias = ba[j * 16 + r];
#pragma unroll
        for (int i = 0; i < 8; ++i) {
            int m = i + half * 8;
            a_out[(rowBase + m) * DIM + j * 16 + r] = tanhf(acc[j][i] + bias);
        }
    }
}

// ---------------------------------------------------------------------------
// Phase 3: serial DKVMN scan. One block per batch, 128 threads = one per
// D-column. Mv column lives in 50 registers per thread; only w goes via LDS.
//   read_t = w_t @ Mv (pre-update);  Mv += w ⊗ (a - Mv*e)
// ---------------------------------------------------------------------------
__global__ __launch_bounds__(128) void dkvmn_recurrence(
    const float* __restrict__ Mv0, const float* __restrict__ w_ws,
    const float* __restrict__ e_ws, const float* __restrict__ a_ws,
    u16* __restrict__ reads_bf)
{
    __shared__ float wlds[64];
    const int b = blockIdx.x;
    const int d = threadIdx.x;          // 0..127

    float mv[MEM];
#pragma unroll
    for (int m = 0; m < MEM; ++m) mv[m] = Mv0[m * DIM + d];

    for (int t = 0; t < SEQT; ++t) {
        const int row = b * SEQT + t;
        if (d < MEM) wlds[d] = w_ws[row * 64 + d];
        __syncthreads();

        const float ev = e_ws[row * DIM + d];
        const float av = a_ws[row * DIM + d];

        float rd = 0.0f;
#pragma unroll
        for (int m = 0; m < MEM; ++m) rd = fmaf(wlds[m], mv[m], rd);
        reads_bf[row * DIM + d] = f2bf(rd);

#pragma unroll
        for (int m = 0; m < MEM; ++m)
            mv[m] = fmaf(wlds[m], av - mv[m] * ev, mv[m]);   // mv*(1-w e)+w a

        __syncthreads();   // protect wlds before next step's refill
    }
}

// ---------------------------------------------------------------------------
// Phase 4: f = tanh([read,k] @ Wf + bf) via WMMA (K=256), fused epilogue:
//   ability = tanh(f@Wab+bab); diff = tanh(k@Wd+bd);
//   out = sigmoid(3*ability - diff)
// ---------------------------------------------------------------------------
__global__ __launch_bounds__(128) void dkvmn_predict(
    const u16* __restrict__ reads_bf, const u16* __restrict__ k_bf,
    const u16* __restrict__ WfT, const float* __restrict__ bfv,
    const float* __restrict__ Wab, const float* __restrict__ bab,
    const float* __restrict__ Wd, const float* __restrict__ bd,
    float* __restrict__ out)
{
    __shared__ float flds[WAVES_PER_BLOCK][16][DIM];

    const int wave = threadIdx.x >> 5;
    const int lane = threadIdx.x & 31;
    const int tile = blockIdx.x * WAVES_PER_BLOCK + wave;
    const int rowBase = tile * 16;
    const int r    = lane & 15;
    const int half = lane >> 4;
    const int row  = rowBase + r;
    const int koff = half * 8;

    f32x8 acc[8];
#pragma unroll
    for (int j = 0; j < 8; ++j) {
        float bias = bfv[j * 16 + r];
#pragma unroll
        for (int i = 0; i < 8; ++i) acc[j][i] = bias;
    }

#pragma unroll
    for (int c = 0; c < 8; ++c) {   // K = 256: chunks 0-3 = read, 4-7 = k
        const u16* src = (c < 4) ? (reads_bf + row * DIM + c * 32)
                                 : (k_bf    + row * DIM + (c - 4) * 32);
        bf16x16 A = make_frag(src + koff, src + koff + 16);
#pragma unroll
        for (int j = 0; j < 8; ++j) {
            const u16* bp = WfT + (j * 16 + r) * 256 + c * 32 + half * 8;
            bf16x16 B = make_frag(bp, bp + 16);
            acc[j] = __builtin_amdgcn_wmma_f32_16x16x32_bf16(
                false, A, false, B, (short)0, acc[j], false, false);
        }
    }

#pragma unroll
    for (int j = 0; j < 8; ++j)
#pragma unroll
        for (int i = 0; i < 8; ++i)
            flds[wave][i + half * 8][j * 16 + r] = tanhf(acc[j][i]);
    __syncthreads();

    if (lane < 16) {
        float ab = 0.0f, qd = 0.0f;
        const u16* kr = k_bf + row * DIM;
        for (int d2 = 0; d2 < DIM; ++d2) {
            ab = fmaf(flds[wave][r][d2], Wab[d2], ab);
            qd = fmaf(bf2f(kr[d2]), Wd[d2], qd);
        }
        float stu = tanhf(ab + bab[0]);
        float dif = tanhf(qd + bd[0]);
        out[row] = sigmoidf(3.0f * stu - dif);
    }
}

// ---------------------------------------------------------------------------
extern "C" void kernel_launch(void* const* d_in, const int* in_sizes, int n_in,
                              void* d_out, int out_size, void* d_ws, size_t ws_size,
                              hipStream_t stream) {
    (void)in_sizes; (void)n_in; (void)out_size; (void)ws_size;

    const int*   concept_seq = (const int*)  d_in[0];
    const int*   correct_seq = (const int*)  d_in[1];
    const float* embed_key   = (const float*)d_in[2];
    const float* embed_value = (const float*)d_in[3];
    const float* Mk          = (const float*)d_in[4];
    const float* Mv0         = (const float*)d_in[5];
    const float* We          = (const float*)d_in[6];
    const float* be          = (const float*)d_in[7];
    const float* Wa          = (const float*)d_in[8];
    const float* ba          = (const float*)d_in[9];
    const float* Wf          = (const float*)d_in[10];
    const float* bfv         = (const float*)d_in[11];
    const float* Wab         = (const float*)d_in[12];
    const float* bab         = (const float*)d_in[13];
    const float* Wd          = (const float*)d_in[14];
    const float* bd          = (const float*)d_in[15];
    float* out = (float*)d_out;

    char* ws = (char*)d_ws;
    u16*   ek   = (u16*)  (ws + OFF_EK);
    u16*   evt  = (u16*)  (ws + OFF_EV);
    u16*   mk   = (u16*)  (ws + OFF_MK);
    u16*   WeT  = (u16*)  (ws + OFF_WET);
    u16*   WaT  = (u16*)  (ws + OFF_WAT);
    u16*   WfT  = (u16*)  (ws + OFF_WFT);
    u16*   k_bf = (u16*)  (ws + OFF_KBF);
    float* w_ws = (float*)(ws + OFF_W);
    float* e_ws = (float*)(ws + OFF_E);
    float* a_ws = (float*)(ws + OFF_A);
    u16*   r_bf = (u16*)  (ws + OFF_RD);

    // Phase 1: bf16 conversions (tiny)
    conv_copy<<<(NUM_CONCEPT*DIM + 255)/256, 256, 0, stream>>>(embed_key, ek, NUM_CONCEPT*DIM);
    conv_copy<<<(2*NUM_CONCEPT*DIM + 255)/256, 256, 0, stream>>>(embed_value, evt, 2*NUM_CONCEPT*DIM);
    conv_mk  <<<(64*DIM)/256, 256, 0, stream>>>(Mk, mk);
    conv_T   <<<(DIM*DIM + 255)/256, 256, 0, stream>>>(We, WeT, DIM, DIM);
    conv_T   <<<(DIM*DIM + 255)/256, 256, 0, stream>>>(Wa, WaT, DIM, DIM);
    conv_T   <<<(2*DIM*DIM + 255)/256, 256, 0, stream>>>(Wf, WfT, 2*DIM, DIM);

    // Phase 2: WMMA gather+GEMMs (2000 tiles, 4 waves/block)
    dkvmn_embed_gemm<<<TILES/WAVES_PER_BLOCK, 128, 0, stream>>>(
        concept_seq, correct_seq, ek, evt, mk, WeT, WaT, be, ba,
        k_bf, w_ws, e_ws, a_ws);

    // Phase 3: serial scan, one block per batch element
    dkvmn_recurrence<<<BATCH, 128, 0, stream>>>(Mv0, w_ws, e_ws, a_ws, r_bf);

    // Phase 4: WMMA output GEMM + fused prediction
    dkvmn_predict<<<TILES/WAVES_PER_BLOCK, 128, 0, stream>>>(
        r_bf, k_bf, WfT, bfv, Wab, bab, Wd, bd, out);
}
